// SequenceLoss_360777253256
// MI455X (gfx1250) — matrix operand order repaired and checked by reading, not compiled, not run on hardware
//
#include <hip/hip_runtime.h>
#include <hip/hip_bf16.h>

typedef __attribute__((ext_vector_type(16))) _Float16 v16h;
typedef __attribute__((ext_vector_type(8)))  float    v8f;

#define BB 32
#define LL 200
#define KK 100
#define DD 64
#define BL (BB * LL)            // 6400 positions
#define WAVES_PER_BLOCK 8       // 256 threads = 8 wave32
#define NTILES 7                // ceil(101 / 16) -> slots 0..111

__global__ void bpr_init(float* ws) {
  if (blockIdx.x == 0 && threadIdx.x < 2) ws[threadIdx.x] = 0.0f;
}

// Score 16 item columns against the broadcast seq vector via 2 WMMAs (K=64).
__device__ __forceinline__ float tile_score(const float* __restrict__ rp, int kb,
                                            const v16h& a0, const v16h& a1) {
  v16h b0, b1;
#pragma unroll
  for (int i = 0; i < 16; ++i) {
    b0[i] = (_Float16)rp[kb + i];        // K-chunk 0: d 0..31
    b1[i] = (_Float16)rp[32 + kb + i];   // K-chunk 1: d 32..63
  }
  v8f c = {};
  c = __builtin_amdgcn_wmma_f32_16x16x32_f16(false, a0, false, b0,
                                             (short)0, c, false, false);
  c = __builtin_amdgcn_wmma_f32_16x16x32_f16(false, a1, false, b1,
                                             (short)0, c, false, false);
  // All A rows identical -> every accumulator element = score of column (lane&15).
  return c[0];
}

__device__ __forceinline__ float bpr_term(float pos, float score) {
  // -log(sigmoid(pos - score) + 1e-8), sigmoid via fast v_rcp_f32.
  const float e  = __expf(score - pos);                 // e^{-diff}
  const float sg = __builtin_amdgcn_rcpf(1.0f + e);
  return -__logf(sg + 1e-8f);
}

__global__ __launch_bounds__(256) void bpr_main(
    const float* __restrict__ seq_embs,    // [BL, 64]
    const float* __restrict__ item_table,  // [1e6, 64]
    const float* __restrict__ mask,        // [BL]
    const int*   __restrict__ target_seq,  // [BL]
    const int*   __restrict__ neg_items,   // [BL, 100]
    float* __restrict__ ws)                // ws[0]=num, ws[1]=den
{
  __shared__ _Float16 sh[WAVES_PER_BLOCK][DD];

  const int tid  = threadIdx.x;
  const int w    = tid >> 5;
  const int lane = tid & 31;
  const int p    = blockIdx.x * WAVES_PER_BLOCK + w;   // position id
  const bool valid = (p < BL);
  const int  pc  = valid ? p : 0;

  // Stage this wave's seq vector into LDS as f16 (64 halves).
  {
    const float2 sv = ((const float2*)(seq_embs + (size_t)pc * DD))[lane];
    sh[w][lane * 2]     = (_Float16)sv.x;
    sh[w][lane * 2 + 1] = (_Float16)sv.y;
  }
  __syncthreads();
  if (!valid) return;

  // A fragments: seq vector broadcast into all 16 rows.
  // A 16x32 k order used here (matches B): lanes<16 hold K=0..15;
  // lanes>=16 hold K=16..31.
  const int hb = (lane & 16) ? 16 : 0;
  v16h a0, a1;
#pragma unroll
  for (int i = 0; i < 8; ++i) {
    a0[i]     = sh[w][hb + i];
    a0[i + 8] = sh[w][8 + hb + i];
    a1[i]     = sh[w][32 + hb + i];
    a1[i + 8] = sh[w][40 + hb + i];
  }

  const int col = lane & 15;               // which item column this lane feeds
  const int kb  = (lane & 16) ? 16 : 0;    // B 32x16: lanes<16 = K 0..15, lanes>=16 = K 16..31
  const float m = mask[p];

  float num = 0.0f;

  // ---- Tile 0 (peeled): slot 0 = positive, slots 1..15 = first negatives ----
  int row0 = (col == 0) ? target_seq[p]
                        : neg_items[(size_t)p * KK + (col - 1)];
  // Resolve + prefetch tile 1's rows before computing tile 0.
  int row_next = neg_items[(size_t)p * KK + (16 + col - 1)];
  {
    const float* pf = item_table + (size_t)row_next * DD + kb;
    __builtin_prefetch(pf, 0, 1);
    __builtin_prefetch(pf + 32, 0, 1);
  }
  float score0 = tile_score(item_table + (size_t)row0 * DD, kb, a0, a1);
  const float pos = __shfl(score0, 0, 32);   // broadcast slot-0 score
  if (col != 0 && lane < 16)
    num += bpr_term(pos, score0);

  // ---- Tiles 1..6 ----
#pragma unroll 1
  for (int t = 1; t < NTILES; ++t) {
    const int  g     = t * 16 + col;            // global slot 16..111
    const bool isneg = (g <= KK);
    const int  row   = row_next;

    if (t + 1 < NTILES) {
      const int gn = (t + 1) * 16 + col;
      const int rn = (gn <= KK) ? neg_items[(size_t)p * KK + (gn - 1)] : 0;
      row_next = rn;
      const float* pf = item_table + (size_t)rn * DD + kb;
      __builtin_prefetch(pf, 0, 1);        // global_prefetch
      __builtin_prefetch(pf + 32, 0, 1);
    }

    const float score = tile_score(item_table + (size_t)row * DD, kb, a0, a1);

    if (isneg && lane < 16)                // lanes 16..31 duplicate columns
      num += bpr_term(pos, score);
  }

  num *= m;
#pragma unroll
  for (int off = 16; off > 0; off >>= 1)
    num += __shfl_down(num, off, 32);

  if (lane == 0) {
    atomicAdd(&ws[0], num);
    atomicAdd(&ws[1], m * (float)KK);
  }
}

__global__ void bpr_final(const float* __restrict__ ws, float* __restrict__ out) {
  if (blockIdx.x == 0 && threadIdx.x == 0) {
    const float den = ws[1];
    out[0] = (den > 0.0f) ? (ws[0] / den) : 0.0f;
  }
}

extern "C" void kernel_launch(void* const* d_in, const int* in_sizes, int n_in,
                              void* d_out, int out_size, void* d_ws, size_t ws_size,
                              hipStream_t stream) {
  const float* seq = (const float*)d_in[0];   // seq_embs  [32,200,64]
  const float* tab = (const float*)d_in[1];   // item_table [1e6,64]
  const float* msk = (const float*)d_in[2];   // mask      [32,200]
  const int*   tgt = (const int*)d_in[3];     // target_seq [32,200]
  const int*   neg = (const int*)d_in[4];     // neg_items [32,200,100]
  float* ws  = (float*)d_ws;
  float* out = (float*)d_out;

  bpr_init<<<1, 64, 0, stream>>>(ws);
  const int blocks = (BL + WAVES_PER_BLOCK - 1) / WAVES_PER_BLOCK;  // 800
  bpr_main<<<blocks, 256, 0, stream>>>(seq, tab, msk, tgt, neg, ws);
  bpr_final<<<1, 32, 0, stream>>>(ws, out);
}
